// MLA_20486994002326
// MI455X (gfx1250) — matrix operand order repaired and checked by reading, not compiled, not run on hardware
//
#include <hip/hip_runtime.h>
#include <hip/hip_bf16.h>

typedef __attribute__((ext_vector_type(16))) _Float16 v16h;
typedef __attribute__((ext_vector_type(8)))  float    v8f;
typedef unsigned int u32x4 __attribute__((ext_vector_type(4)));
typedef int          i32x8 __attribute__((ext_vector_type(8)));
typedef int          i32x4 __attribute__((ext_vector_type(4)));

#define H_    8
#define VD_   32
#define QKP_  64      // QKD=48 padded to 64
#define B_    2
#define N_    4096
#define D_    512
#define NT_   (B_*N_) // 8192 rows

#if __has_include(<hip/amd_detail/amd_gfx1250_TDM.h>)
#define TDM_6ARG 1
#endif

// ---------------------------------------------------------------------------
// TDM: issue a 2D tile load (16-bit elements) Global -> LDS. D# per ISA §8.
//   lds_off       : LDS byte address of tile destination
//   gaddr         : global byte address of tile start
//   tile_d0/d1    : tile extent (elements) along contiguous / strided dim
//   tensor_d0/d1  : tensor extent for OOB clamping (elements)
//   d0_stride     : elements between consecutive dim-1 lines
// ---------------------------------------------------------------------------
#if __has_builtin(__builtin_amdgcn_tensor_load_to_lds)
#define HAVE_TDM 1
__device__ __forceinline__ void tdm_load_tile_2d_f16(
    unsigned lds_off, const void* gaddr,
    unsigned tile_d0, unsigned tile_d1,
    unsigned tensor_d0, unsigned tensor_d1,
    unsigned long long d0_stride)
{
  unsigned long long ga = (unsigned long long)(uintptr_t)gaddr;
  u32x4 g0;
  g0[0] = 1u;                                         // count=1, user-mode
  g0[1] = lds_off;                                    // lds_addr
  g0[2] = (unsigned)(ga & 0xffffffffu);               // global_addr[31:0]
  g0[3] = (unsigned)((ga >> 32) & 0x01ffffffu)        // global_addr[56:32]
        | (2u << 30);                                 // type = 2 ("image")
  i32x8 g1;
  g1[0] = (int)(1u << 16);                            // data_size=1 (2B), wg_mask=0
  g1[1] = (int)((tensor_d0 & 0xffffu) << 16);         // tensor_dim0[15:0] -> bits[63:48]
  g1[2] = (int)((tensor_d0 >> 16) | ((tensor_d1 & 0xffffu) << 16));
  g1[3] = (int)((tensor_d1 >> 16) | ((tile_d0 & 0xffffu) << 16));
  g1[4] = (int)(tile_d1 & 0xffffu);                   // tile_dim1; tile_dim2=0
  g1[5] = (int)(d0_stride & 0xffffffffull);           // dim0_stride[31:0]
  g1[6] = (int)((d0_stride >> 32) & 0xffffull);       // dim0_stride[47:32]; dim1_stride=0
  g1[7] = 0;
  i32x4 g2 = {0, 0, 0, 0};
  i32x4 g3 = {0, 0, 0, 0};
#ifdef TDM_6ARG
  i32x8 g4 = {0, 0, 0, 0, 0, 0, 0, 0};
  __builtin_amdgcn_tensor_load_to_lds(g0, g1, g2, g3, g4, 0);
#else
  __builtin_amdgcn_tensor_load_to_lds(g0, g1, g2, g3, 0);
#endif
}
#endif

// ---------------------------------------------------------------------------
// fp32-in / fp32-out GEMM, f16 WMMA compute. C[M,Nn] = A[M,K] @ B[K,Nn]
// One wave -> one 16x16 C tile; 16-row A tile staged once in LDS per block.
// Requires M%16==0, K%32==0, Nn%16==0.
// ---------------------------------------------------------------------------
__global__ __launch_bounds__(256) void gemm_wmma_f16(
    const float* __restrict__ A, const float* __restrict__ B,
    float* __restrict__ C, int M, int Kdim, int Nn)
{
  __shared__ _Float16 As[16][32];
  const int lane = threadIdx.x & 31;
  const int wave = threadIdx.x >> 5;
  const int m0   = blockIdx.x * 16;
  const int n0   = (blockIdx.y * 8 + wave) * 16;
  const int hl   = lane >> 4;
  const int l15  = lane & 15;
  const int col  = n0 + l15;
  const bool active = (n0 < Nn);

  v8f acc = {};
  for (int k0 = 0; k0 < Kdim; k0 += 32) {
    __syncthreads();
    for (int i = threadIdx.x; i < 16 * 32; i += 256) {
      int r = i >> 5, k = i & 31;
      As[r][k] = (_Float16)A[(size_t)(m0 + r) * Kdim + k0 + k];
    }
    __syncthreads();
    if (active) {
      v16h a, b;
      const _Float16* ar = &As[l15][hl * 16];
#pragma unroll
      for (int e = 0; e < 8; ++e) { a[e] = ar[e]; a[8 + e] = ar[8 + e]; }
      const float* bcol = B + (size_t)(k0 + hl * 16) * Nn + col;
#pragma unroll
      for (int e = 0; e < 16; ++e) b[e] = (_Float16)bcol[(size_t)e * Nn];
      acc = __builtin_amdgcn_wmma_f32_16x16x32_f16(false, a, false, b, (short)0, acc, false, false);
    }
  }
  if (active) {
#pragma unroll
    for (int j = 0; j < 8; ++j)
      C[(size_t)(m0 + j + hl * 8) * Nn + col] = acc[j];
  }
}

// ---------------------------------------------------------------------------
// RoPE(q) -> padded f16 Q; RoPE(kv[:16]) -> f16 k_rope; RMSNorm(kv[16:]) -> kv_lat
// ---------------------------------------------------------------------------
__global__ __launch_bounds__(256) void prep_kernel(
    const float* __restrict__ q_raw, const float* __restrict__ kv_raw,
    const float* __restrict__ cosb, const float* __restrict__ sinb,
    const float* __restrict__ kvw,
    _Float16* __restrict__ q16, _Float16* __restrict__ krope,
    float* __restrict__ kv_lat)
{
  int row = blockIdx.x * blockDim.x + threadIdx.x;
  if (row >= NT_) return;
  int n = row & (N_ - 1);
  for (int h = 0; h < H_; ++h) {
    const float* qs = q_raw + (size_t)row * 384 + h * 48;
    _Float16*    qd = q16 + ((size_t)row * H_ + h) * QKP_;
    for (int j = 0; j < 32; ++j) qd[j] = (_Float16)qs[j];
    for (int d = 0; d < 8; ++d) {
      float c = cosb[n * 8 + d], s = sinb[n * 8 + d];
      float x0 = qs[32 + 2 * d], x1 = qs[32 + 2 * d + 1];
      qd[32 + 2 * d]     = (_Float16)(x0 * c - x1 * s);
      qd[32 + 2 * d + 1] = (_Float16)(x0 * s + x1 * c);
    }
    for (int j = 48; j < 64; ++j) qd[j] = (_Float16)0.f;
  }
  const float* kvs = kv_raw + (size_t)row * 144;
  for (int d = 0; d < 8; ++d) {
    float c = cosb[n * 8 + d], s = sinb[n * 8 + d];
    float x0 = kvs[2 * d], x1 = kvs[2 * d + 1];
    krope[(size_t)row * 16 + 2 * d]     = (_Float16)(x0 * c - x1 * s);
    krope[(size_t)row * 16 + 2 * d + 1] = (_Float16)(x0 * s + x1 * c);
  }
  float ss = 0.f;
  for (int i = 0; i < 128; ++i) { float v = kvs[16 + i]; ss += v * v; }
  float inv = rsqrtf(ss * (1.f / 128.f) + 1e-6f);
  for (int i = 0; i < 128; ++i) kv_lat[(size_t)row * 128 + i] = kvs[16 + i] * inv * kvw[i];
}

// ---------------------------------------------------------------------------
// Assemble K = [k_nope | k_rope | 0-pad] and V per head, f16
// ---------------------------------------------------------------------------
__global__ __launch_bounds__(256) void assemble_kernel(
    const float* __restrict__ kvb, const _Float16* __restrict__ krope,
    _Float16* __restrict__ k16, _Float16* __restrict__ v16p)
{
  int idx = blockIdx.x * blockDim.x + threadIdx.x; // row*H + h
  if (idx >= NT_ * H_) return;
  int row = idx >> 3, h = idx & 7;
  const float* src = kvb + (size_t)row * 512 + h * 64;
  _Float16* kd = k16 + (size_t)idx * QKP_;
  _Float16* vd = v16p + (size_t)idx * VD_;
  for (int j = 0; j < 32; ++j) kd[j] = (_Float16)src[j];
  for (int j = 0; j < 16; ++j) kd[32 + j] = krope[(size_t)row * 16 + j];
  for (int j = 48; j < 64; ++j) kd[j] = (_Float16)0.f;
  for (int j = 0; j < 32; ++j) vd[j] = (_Float16)src[32 + j];
}

// ---------------------------------------------------------------------------
// Flash attention: 8 waves/block, wave = 16 q-rows, K/V tiles of 32 in LDS.
// K tile fetched by the Tensor Data Mover (TDM); V cooperatively transposed.
// ---------------------------------------------------------------------------
#define SM_K_BYTES (32 * QKP_ * 2)       // 4 KB, at LDS offset 0
#define SM_V_BYTES (VD_ * 32 * 2)        // 2 KB
#define SM_P_BYTES (8 * 16 * 32 * 2)     // 8 KB

__global__ __launch_bounds__(256) void flash_kernel(
    const _Float16* __restrict__ Q, const _Float16* __restrict__ Kk,
    const _Float16* __restrict__ V, float* __restrict__ Oout)
{
  __shared__ __align__(128) char smem[SM_K_BYTES + SM_V_BYTES + SM_P_BYTES];
  _Float16 (*Kt)[QKP_]    = (_Float16(*)[QKP_])(smem);
  _Float16 (*Vt)[32]      = (_Float16(*)[32])(smem + SM_K_BYTES);
  _Float16 (*Pst)[16][32] = (_Float16(*)[16][32])(smem + SM_K_BYTES + SM_V_BYTES);

  const int lane = threadIdx.x & 31;
  const int wave = threadIdx.x >> 5;
  const int b    = blockIdx.y >> 3;
  const int h    = blockIdx.y & 7;
  const int q0   = blockIdx.x * 128 + wave * 16;
  const int hl   = lane >> 4;
  const int l15  = lane & 15;
  const float scale = 0.14433756729740643f; // 48^-0.5

  // Q A-fragments (rows q0+l15, K chunks [0,32) and [32,64))
  const _Float16* qrow = Q + (((size_t)(b * N_) + q0 + l15) * H_ + h) * QKP_;
  v16h aq0, aq1;
#pragma unroll
  for (int e = 0; e < 8; ++e) {
    aq0[e]     = qrow[hl * 16 + e];
    aq0[8 + e] = qrow[hl * 16 + 8 + e];
    aq1[e]     = qrow[32 + hl * 16 + e];
    aq1[8 + e] = qrow[32 + hl * 16 + 8 + e];
  }

  float mrow[8], lrow[8];
  v8f O0 = {}, O1 = {};
#pragma unroll
  for (int j = 0; j < 8; ++j) { mrow[j] = -1e30f; lrow[j] = 0.f; }

  int kend = blockIdx.x * 128 + 128;  // causal bound for this block
  if (kend > N_) kend = N_;

  for (int kt = 0; kt < kend; kt += 32) {
    // --- K tile: 32 rows x 64 cols of f16, row stride H_*QKP_ elements ---
#if defined(HAVE_TDM)
    if (wave == 0) {
      const void* gk = Kk + (((size_t)(b * N_) + kt) * H_ + h) * QKP_;
      tdm_load_tile_2d_f16(/*lds_off=*/0u, gk,
                           /*tile_d0=*/QKP_, /*tile_d1=*/32,
                           /*tensor_d0=*/QKP_, /*tensor_d1=*/32,
                           /*d0_stride=*/(unsigned long long)(H_ * QKP_));
    }
#else
    for (int i = threadIdx.x; i < 32 * QKP_; i += 256) {
      int kr = i >> 6, kc = i & 63;
      Kt[kr][kc] = Kk[(((size_t)(b * N_) + kt + kr) * H_ + h) * QKP_ + kc];
    }
#endif
    // --- V tile, transposed into [vd][k] ---
    for (int i = threadIdx.x; i < 32 * VD_; i += 256) {
      int kr = i >> 5, vd = i & 31;
      Vt[vd][kr] = V[(((size_t)(b * N_) + kt + kr) * H_ + h) * VD_ + vd];
    }
#if defined(HAVE_TDM)
    if (wave == 0) __builtin_amdgcn_s_wait_tensorcnt(0);
#endif
    __syncthreads();

    if (kt <= q0 + 15) {  // wave participates (causal)
      // S = Q K^T : two 16-col chunks, each accumulated over K=64 (2 WMMAs)
      v8f Sa = {}, Sb = {};
      {
        const _Float16* kpa = &Kt[l15][0];
        const _Float16* kpb = &Kt[16 + l15][0];
        v16h b00 = *(const v16h*)(kpa + hl * 16);
        v16h b01 = *(const v16h*)(kpa + 32 + hl * 16);
        v16h b10 = *(const v16h*)(kpb + hl * 16);
        v16h b11 = *(const v16h*)(kpb + 32 + hl * 16);
        Sa = __builtin_amdgcn_wmma_f32_16x16x32_f16(false, aq0, false, b00, (short)0, Sa, false, false);
        Sa = __builtin_amdgcn_wmma_f32_16x16x32_f16(false, aq1, false, b01, (short)0, Sa, false, false);
        Sb = __builtin_amdgcn_wmma_f32_16x16x32_f16(false, aq0, false, b10, (short)0, Sb, false, false);
        Sb = __builtin_amdgcn_wmma_f32_16x16x32_f16(false, aq1, false, b11, (short)0, Sb, false, false);
      }
      // online softmax (C layout: row = j + 8*hl, col = l15 / 16+l15)
      const int ka = kt + l15, kb = kt + 16 + l15;
#pragma unroll
      for (int j = 0; j < 8; ++j) {
        const int rq = q0 + j + hl * 8;
        float sa = Sa[j] * scale + ((ka > rq) ? -1e9f : 0.f);
        float sb = Sb[j] * scale + ((kb > rq) ? -1e9f : 0.f);
        float mj = fmaxf(sa, sb);
        mj = fmaxf(mj, __shfl_xor(mj, 1));
        mj = fmaxf(mj, __shfl_xor(mj, 2));
        mj = fmaxf(mj, __shfl_xor(mj, 4));
        mj = fmaxf(mj, __shfl_xor(mj, 8));
        float mnew = fmaxf(mrow[j], mj);
        float pa = __expf(sa - mnew);
        float pb = __expf(sb - mnew);
        float rs = pa + pb;
        rs += __shfl_xor(rs, 1);
        rs += __shfl_xor(rs, 2);
        rs += __shfl_xor(rs, 4);
        rs += __shfl_xor(rs, 8);
        float corr = __expf(mrow[j] - mnew);
        lrow[j] = lrow[j] * corr + rs;
        mrow[j] = mnew;
        O0[j] *= corr;
        O1[j] *= corr;
        Pst[wave][j + hl * 8][l15]      = (_Float16)pa;
        Pst[wave][j + hl * 8][16 + l15] = (_Float16)pb;
      }
      asm volatile("s_wait_dscnt 0" ::: "memory"); // wave-local LDS RAW fence
      // reload P in A layout (rows l15, K = k-position in tile)
      v16h aP;
      {
        const _Float16* pr = &Pst[wave][l15][0];
#pragma unroll
        for (int e = 0; e < 8; ++e) {
          aP[e]     = pr[hl * 16 + e];
          aP[8 + e] = pr[hl * 16 + 8 + e];
        }
      }
      // O += P V  (V^T in LDS -> contiguous B fragments)
      v16h bv0 = *(const v16h*)(&Vt[l15][0]      + hl * 16);
      v16h bv1 = *(const v16h*)(&Vt[16 + l15][0] + hl * 16);
      O0 = __builtin_amdgcn_wmma_f32_16x16x32_f16(false, aP, false, bv0, (short)0, O0, false, false);
      O1 = __builtin_amdgcn_wmma_f32_16x16x32_f16(false, aP, false, bv1, (short)0, O1, false, false);
    }
    __syncthreads();
  }

  // normalize + write [b, n, h*VD]
#pragma unroll
  for (int j = 0; j < 8; ++j) {
    const int r = q0 + j + hl * 8;
    const float inv = 1.f / lrow[j];
    float* dst = Oout + ((size_t)(b * N_) + r) * (H_ * VD_) + h * VD_;
    dst[l15]      = O0[j] * inv;
    dst[16 + l15] = O1[j] * inv;
  }
}

// ---------------------------------------------------------------------------
extern "C" void kernel_launch(void* const* d_in, const int* in_sizes, int n_in,
                              void* d_out, int out_size, void* d_ws, size_t ws_size,
                              hipStream_t stream) {
  const float* x     = (const float*)d_in[0];
  const float* cosb  = (const float*)d_in[2];
  const float* sinb  = (const float*)d_in[3];
  const float* Wq    = (const float*)d_in[5];
  const float* Wkv_a = (const float*)d_in[6];
  const float* kvw   = (const float*)d_in[7];
  const float* Wkv_b = (const float*)d_in[8];
  const float* Wout  = (const float*)d_in[9];
  float* out = (float*)d_out;

  char* ws = (char*)d_ws;
  size_t off = 0;
  auto alloc = [&](size_t bytes) { char* p = ws + off; off += (bytes + 255) & ~(size_t)255; return p; };
  float*    q_raw  = (float*)   alloc((size_t)NT_ * 384 * 4);
  float*    kv_raw = (float*)   alloc((size_t)NT_ * 144 * 4);
  _Float16* q16    = (_Float16*)alloc((size_t)NT_ * H_ * QKP_ * 2);
  float*    kv_lat = (float*)   alloc((size_t)NT_ * 128 * 4);
  _Float16* krope  = (_Float16*)alloc((size_t)NT_ * 16 * 2);
  float*    kvb    = (float*)   alloc((size_t)NT_ * 512 * 4);
  _Float16* k16    = (_Float16*)alloc((size_t)NT_ * H_ * QKP_ * 2);
  _Float16* v16p   = (_Float16*)alloc((size_t)NT_ * H_ * VD_ * 2);
  float*    attn   = (float*)   alloc((size_t)NT_ * (H_ * VD_) * 4);

  dim3 blk(256);
  gemm_wmma_f16<<<dim3(NT_ / 16, 3), blk, 0, stream>>>(x, Wq, q_raw, NT_, D_, 384);
  gemm_wmma_f16<<<dim3(NT_ / 16, 2), blk, 0, stream>>>(x, Wkv_a, kv_raw, NT_, D_, 144);
  prep_kernel<<<dim3(NT_ / 256), blk, 0, stream>>>(q_raw, kv_raw, cosb, sinb, kvw,
                                                   q16, krope, kv_lat);
  gemm_wmma_f16<<<dim3(NT_ / 16, 4), blk, 0, stream>>>(kv_lat, Wkv_b, kvb, NT_, 128, 512);
  assemble_kernel<<<dim3(NT_ * H_ / 256), blk, 0, stream>>>(kvb, krope, k16, v16p);
  flash_kernel<<<dim3(N_ / 128, B_ * H_), blk, 0, stream>>>(q16, k16, v16p, attn);
  gemm_wmma_f16<<<dim3(NT_ / 16, 4), blk, 0, stream>>>(attn, Wout, out, NT_, 256, 512);
}
